// SelfAttention_14310831030812
// MI455X (gfx1250) — matrix operand order repaired and checked by reading, not compiled, not run on hardware
//
#include <hip/hip_runtime.h>
#include <hip/hip_bf16.h>

typedef __attribute__((ext_vector_type(16))) __bf16 bf16x16;
typedef __attribute__((ext_vector_type(8)))  float  f32x8;
typedef __attribute__((ext_vector_type(4)))  unsigned int u32x4;
typedef __attribute__((ext_vector_type(8)))  int  i32x8;
typedef __attribute__((ext_vector_type(4)))  int  i32x4;

union Frag { uint4 u4[2]; bf16x16 v; };

__device__ __forceinline__ __bf16 f2bf(float f) {
    unsigned u = __float_as_uint(f);
    u += 0x7FFFu + ((u >> 16) & 1u);            // round-to-nearest-even
    unsigned short h = (unsigned short)(u >> 16);
    __bf16 r;
    __builtin_memcpy(&r, &h, 2);
    return r;
}

#define WMMA_BF16(a, b, c) \
    __builtin_amdgcn_wmma_f32_16x16x32_bf16(false, (a), false, (b), (short)0, (c), false, false)

constexpr int BM = 128, BN = 128, BK = 32, LDST = 40; // 80B LDS row stride (16B aligned)

// ---------------------------------------------------------------------------
// Tensor Data Mover: DMA a 2D bf16 tile [tile_rows x BK] from a [rows_total x
// row_len] row-major tensor into LDS, padding each 64B row to 80B (LDST=40).
// D# packing per CDNA5 ISA.
// ---------------------------------------------------------------------------
__device__ __forceinline__ void tdm_load_tile(unsigned lds_off, const __bf16* gptr,
                                              unsigned row_len, unsigned rows_total,
                                              unsigned tile_rows) {
    unsigned long long ga = (unsigned long long)(const void*)gptr;
    u32x4 g0;
    g0[0] = 1u;                                   // count=1 (valid), user mode
    g0[1] = lds_off;                              // lds_addr (bytes)
    g0[2] = (unsigned)(ga & 0xFFFFFFFFu);         // global_addr[31:0]
    g0[3] = (unsigned)((ga >> 32) & 0x1FFFFFFu)   // global_addr[56:32]
          | (2u << 30);                           // type=2 ("image")
    i32x8 g1;
    // data_size=1(2B) | pad_enable | pad_interval=3(16 DW) | pad_amount=3(4 DW)
    g1[0] = (int)((1u << 16) | (1u << 20) | (3u << 22) | (3u << 25));
    g1[1] = (int)((row_len & 0xFFFFu) << 16);                      // tensor_dim0 lo16
    g1[2] = (int)(((row_len >> 16) & 0xFFFFu)
          | ((rows_total & 0xFFFFu) << 16));                       // td0 hi16 | td1 lo16
    g1[3] = (int)(((rows_total >> 16) & 0xFFFFu)
          | ((unsigned)BK << 16));                                 // td1 hi16 | tile_dim0
    g1[4] = (int)(tile_rows & 0xFFFFu);                            // tile_dim1, tile_dim2=0
    g1[5] = (int)row_len;                                          // tensor_dim0_stride lo32
    g1[6] = 0;                                                     // stride0 hi16, stride1 lo16
    g1[7] = 0;
    i32x4 gz = {0, 0, 0, 0};
#if defined(__clang_major__) && (__clang_major__ >= 23)
    i32x8 gz8 = {0, 0, 0, 0, 0, 0, 0, 0};
    __builtin_amdgcn_tensor_load_to_lds(g0, g1, gz, gz, gz8, 0);
#else
    __builtin_amdgcn_tensor_load_to_lds(g0, g1, gz, gz, 0);
#endif
}

__device__ __forceinline__ unsigned lds_offset_of(const void* p) {
    return (unsigned)(unsigned long long)p;       // low 32 bits of flat LDS address
}

// ---------------------------------------------------------------------------
// Prep: f32 -> bf16 copy (for x), and f32 [K][N] -> bf16 [N][K] transpose
// (for weights), so GEMM tiles are pure TDM byte-copies.
// ---------------------------------------------------------------------------
__global__ __launch_bounds__(256) void cvt_kernel(const float* __restrict__ src,
                                                  __bf16* __restrict__ dst) {
    size_t i = ((size_t)blockIdx.x * 256 + threadIdx.x) * 4;
    const float4 v = *(const float4*)(src + i);
    union { __bf16 h[4]; uint2 u; } r;
    r.h[0] = f2bf(v.x); r.h[1] = f2bf(v.y); r.h[2] = f2bf(v.z); r.h[3] = f2bf(v.w);
    *(uint2*)(dst + i) = r.u;
}

__global__ __launch_bounds__(256) void transpose_cvt_kernel(const float* __restrict__ W,
                                                            __bf16* __restrict__ Wt,
                                                            int K, int N) {
    __shared__ float tile[32][33];
    const int n0 = blockIdx.x * 32, k0 = blockIdx.y * 32;
    const int tx = threadIdx.x, ty = threadIdx.y;      // 32 x 8
    for (int r = 0; r < 4; r++)
        tile[ty * 4 + r][tx] = W[(size_t)(k0 + ty * 4 + r) * N + n0 + tx];
    __syncthreads();
    for (int r = 0; r < 4; r++)
        Wt[(size_t)(n0 + ty * 4 + r) * K + k0 + tx] = f2bf(tile[tx][ty * 4 + r]);
}

// ---------------------------------------------------------------------------
// GEMM: C[M,N] = A[M,K](bf16) * Wt[N,K](bf16)^T + bias, TDM-staged, double-
// buffered LDS, 8 waves, 128x128 tile, K-step 32. Waves 2(M) x 4(N).
// MODE 0: scatter Q(bf16,[b,h,t,d], *0.125), K(bf16), V^T(bf16,[b,h,d,t])
// MODE 1: f32 out + bias
// ---------------------------------------------------------------------------
template <int MODE>
__global__ __launch_bounds__(256) void gemm_kernel(
    const __bf16* __restrict__ A, const __bf16* __restrict__ Wt,
    const float* __restrict__ bias,
    __bf16* __restrict__ Qb, __bf16* __restrict__ Kb, __bf16* __restrict__ Vt,
    float* __restrict__ Out, int M, int N, int K)
{
    __shared__ __align__(16) __bf16 sA[2][BM * LDST];
    __shared__ __align__(16) __bf16 sB[2][BN * LDST];

    const int tid  = threadIdx.x;
    const int lane = tid & 31;
    const int wave = tid >> 5;
    const int wm   = wave >> 2;      // 0..1
    const int wn   = wave & 3;       // 0..3
    const int lm   = lane & 15;
    const int lh   = lane >> 4;
    const int bn0  = blockIdx.x * BN;
    const int bm0  = blockIdx.y * BM;

    f32x8 acc[4][2];
    for (int i = 0; i < 4; i++)
        for (int j = 0; j < 2; j++)
            for (int e = 0; e < 8; e++) acc[i][j][e] = 0.f;

    const int nk = K / BK;
    // Prologue: wave 0 issues TDM DMA for tile 0 (EXEC-independent, wave-level).
    if (wave == 0) {
        tdm_load_tile(lds_offset_of(sA[0]), A  + (size_t)bm0 * K, K, M, BM);
        tdm_load_tile(lds_offset_of(sB[0]), Wt + (size_t)bn0 * K, K, N, BN);
    }

    for (int ki = 0; ki < nk; ki++) {
        const int cur = ki & 1, nxt = cur ^ 1;
        if (wave == 0) {
            if (ki + 1 < nk) {
                // prefetch next tile pair, then retire the two oldest DMAs (tile ki)
                tdm_load_tile(lds_offset_of(sA[nxt]), A  + (size_t)bm0 * K + (ki + 1) * BK, K, M, BM);
                tdm_load_tile(lds_offset_of(sB[nxt]), Wt + (size_t)bn0 * K + (ki + 1) * BK, K, N, BN);
                __builtin_amdgcn_s_wait_tensorcnt(2);
            } else {
                __builtin_amdgcn_s_wait_tensorcnt(0);
            }
        }
        __syncthreads();                  // publish TDM-written LDS to all waves

        Frag bfr[2];
        for (int ns = 0; ns < 2; ns++) {
            const __bf16* p = sB[cur] + (wn * 32 + ns * 16 + lm) * LDST + lh * 16;
            bfr[ns].u4[0] = *(const uint4*)(p);
            bfr[ns].u4[1] = *(const uint4*)(p + 8);
        }
        for (int ms = 0; ms < 4; ms++) {
            Frag a;
            const __bf16* p = sA[cur] + (wm * 64 + ms * 16 + lm) * LDST + lh * 16;
            a.u4[0] = *(const uint4*)(p);
            a.u4[1] = *(const uint4*)(p + 8);
            for (int ns = 0; ns < 2; ns++)
                acc[ms][ns] = WMMA_BF16(a.v, bfr[ns].v, acc[ms][ns]);
        }
        __syncthreads();                  // all waves done reading before buf reuse
    }

    for (int ms = 0; ms < 4; ms++) {
        for (int ns = 0; ns < 2; ns++) {
            int n = bn0 + wn * 32 + ns * 16 + lm;
            float bv = bias[n];
            for (int i = 0; i < 8; i++) {
                int m = bm0 + wm * 64 + ms * 16 + i + 8 * lh;
                float val = acc[ms][ns][i] + bv;
                if (MODE == 0) {
                    int region = n >> 10, cc = n & 1023;
                    int h = cc >> 6, d = cc & 63;
                    int b_ = m >> 9, t = m & 511;
                    size_t bh = (size_t)b_ * 16 + h;
                    if (region == 0)      Qb[(bh * 512 + t) * 64 + d] = f2bf(val * 0.125f);
                    else if (region == 1) Kb[(bh * 512 + t) * 64 + d] = f2bf(val);
                    else                  Vt[(bh * 64 + d) * 512 + t] = f2bf(val);
                } else {
                    Out[(size_t)m * N + n] = val;
                }
            }
        }
    }
}

// ---------------------------------------------------------------------------
// Flash attention: 1 wave = 16 Q rows, online softmax over 32-key steps.
// Q,K row-major [bh][t][d] bf16; V transposed [bh][d][t] bf16.
// Output O bf16 [b][t][h][d] (== [b][t][c], ready for out-proj GEMM).
// ---------------------------------------------------------------------------
__global__ __launch_bounds__(128) void attn_kernel(
    const __bf16* __restrict__ Qb, const __bf16* __restrict__ Kb,
    const __bf16* __restrict__ Vt, __bf16* __restrict__ Ob)
{
    __shared__ __align__(16) __bf16 sP[4 * 16 * LDST];   // per-wave P transpose buffer

    const int lane = threadIdx.x & 31;
    const int wave = threadIdx.x >> 5;
    const int lm = lane & 15, lh = lane >> 4;
    const int bh = blockIdx.y;                           // b*16 + h
    const int qb = blockIdx.x * 64 + wave * 16;

    const __bf16* Qp = Qb + (size_t)bh * 512 * 64;
    const __bf16* Kp = Kb + (size_t)bh * 512 * 64;
    const __bf16* Vp = Vt + (size_t)bh * 64 * 512;
    __bf16* ldsP = sP + wave * 16 * LDST;

    Frag qf[2];
    {
        const __bf16* p = Qp + (size_t)(qb + lm) * 64 + lh * 16;
        qf[0].u4[0] = *(const uint4*)(p);
        qf[0].u4[1] = *(const uint4*)(p + 8);
        qf[1].u4[0] = *(const uint4*)(p + 32);
        qf[1].u4[1] = *(const uint4*)(p + 40);
    }

    float mst[8], lst[8];
    f32x8 o[4];
    for (int i = 0; i < 8; i++) { mst[i] = -3.0e38f; lst[i] = 0.f; }
    for (int ns = 0; ns < 4; ns++)
        for (int i = 0; i < 8; i++) o[ns][i] = 0.f;

    const int kend = qb + 16;                            // causal bound
    for (int kt0 = 0; kt0 < kend; kt0 += 32) {
        f32x8 sc[2];
        for (int st = 0; st < 2; st++) {
            for (int e = 0; e < 8; e++) sc[st][e] = 0.f;
            const __bf16* kp = Kp + (size_t)(kt0 + st * 16 + lm) * 64 + lh * 16;
            Frag b0, b1;
            b0.u4[0] = *(const uint4*)(kp);      b0.u4[1] = *(const uint4*)(kp + 8);
            b1.u4[0] = *(const uint4*)(kp + 32); b1.u4[1] = *(const uint4*)(kp + 40);
            sc[st] = WMMA_BF16(qf[0].v, b0.v, sc[st]);
            sc[st] = WMMA_BF16(qf[1].v, b1.v, sc[st]);
        }
        for (int st = 0; st < 2; st++) {
            int ncol = kt0 + st * 16 + lm;
            for (int i = 0; i < 8; i++) {
                int row = qb + i + 8 * lh;
                if (ncol > row) sc[st][i] = -3.0e38f;
            }
        }
        float mcur[8];
        for (int i = 0; i < 8; i++) mcur[i] = fmaxf(sc[0][i], sc[1][i]);
        for (int off = 1; off < 16; off <<= 1)
            for (int i = 0; i < 8; i++) mcur[i] = fmaxf(mcur[i], __shfl_xor(mcur[i], off, 16));
        float mnew[8], scl[8], ladd[8];
        for (int i = 0; i < 8; i++) {
            mnew[i] = fmaxf(mst[i], mcur[i]);
            scl[i]  = __expf(mst[i] - mnew[i]);
            float p0 = __expf(sc[0][i] - mnew[i]);
            float p1 = __expf(sc[1][i] - mnew[i]);
            sc[0][i] = p0; sc[1][i] = p1;
            ladd[i] = p0 + p1;
        }
        for (int off = 1; off < 16; off <<= 1)
            for (int i = 0; i < 8; i++) ladd[i] += __shfl_xor(ladd[i], off, 16);
        for (int i = 0; i < 8; i++) { lst[i] = lst[i] * scl[i] + ladd[i]; mst[i] = mnew[i]; }
        for (int ns = 0; ns < 4; ns++)
            for (int i = 0; i < 8; i++) o[ns][i] *= scl[i];

        // reshape P (C-layout) -> A-fragment via per-wave LDS (DS ops in-order)
        for (int st = 0; st < 2; st++)
            for (int i = 0; i < 8; i++)
                ldsP[(i + 8 * lh) * LDST + st * 16 + lm] = f2bf(sc[st][i]);
        Frag pf;
        {
            const __bf16* p = ldsP + lm * LDST + lh * 16;
            pf.u4[0] = *(const uint4*)(p);
            pf.u4[1] = *(const uint4*)(p + 8);
        }
        for (int ns = 0; ns < 4; ns++) {
            const __bf16* vp = Vp + (size_t)(ns * 16 + lm) * 512 + kt0 + lh * 16;
            Frag vf;
            vf.u4[0] = *(const uint4*)(vp);
            vf.u4[1] = *(const uint4*)(vp + 8);
            o[ns] = WMMA_BF16(pf.v, vf.v, o[ns]);
        }
    }

    const int b_ = bh >> 4, h = bh & 15;
    for (int ns = 0; ns < 4; ns++) {
        for (int i = 0; i < 8; i++) {
            int t = qb + i + 8 * lh;
            int d = ns * 16 + lm;
            float val = o[ns][i] / lst[i];
            Ob[(((size_t)b_ * 512 + t) * 16 + h) * 64 + d] = f2bf(val);
        }
    }
}

// ---------------------------------------------------------------------------
extern "C" void kernel_launch(void* const* d_in, const int* in_sizes, int n_in,
                              void* d_out, int out_size, void* d_ws, size_t ws_size,
                              hipStream_t stream) {
    const float* x     = (const float*)d_in[0];   // [32,512,1024]
    const float* w_qkv = (const float*)d_in[1];   // [1024,3072]
    const float* b_qkv = (const float*)d_in[2];   // [3072]
    const float* w_out = (const float*)d_in[3];   // [1024,1024]
    const float* b_out = (const float*)d_in[4];   // [1024]
    float* out = (float*)d_out;                   // [32,512,1024]

    const size_t NE = (size_t)32 * 16 * 512 * 64;       // 16,777,216 elems / buffer
    char* ws = (char*)d_ws;
    __bf16* Qb   = (__bf16*)(ws);                       // 32 MiB
    __bf16* Kb   = (__bf16*)(ws + 2 * NE);              // 32 MiB
    __bf16* Vt   = (__bf16*)(ws + 4 * NE);              // 32 MiB
    __bf16* Ob   = (__bf16*)(ws + 6 * NE);              // 32 MiB
    __bf16* Xb   = (__bf16*)(ws + 8 * NE);              // 32 MiB
    __bf16* Wqt  = (__bf16*)(ws + 10 * NE);             //  6 MiB  [3072][1024]
    __bf16* Wot  = (__bf16*)(ws + 10 * NE + 2 * (size_t)3072 * 1024);  // 2 MiB [1024][1024]

    // 0) one-shot precision/layout prep (memory-trivial: ~110 MB @ 23.3 TB/s)
    cvt_kernel<<<dim3((unsigned)(NE / 1024)), 256, 0, stream>>>(x, Xb);
    transpose_cvt_kernel<<<dim3(3072 / 32, 1024 / 32), dim3(32, 8), 0, stream>>>(w_qkv, Wqt, 1024, 3072);
    transpose_cvt_kernel<<<dim3(1024 / 32, 1024 / 32), dim3(32, 8), 0, stream>>>(w_out, Wot, 1024, 1024);

    // 1) QKV projection: [16384,1024] x [1024,3072] (TDM-staged, double-buffered)
    gemm_kernel<0><<<dim3(3072 / BN, 16384 / BM), 256, 0, stream>>>(
        Xb, Wqt, b_qkv, Qb, Kb, Vt, nullptr, 16384, 3072, 1024);

    // 2) causal flash attention: grid (T/64, B*H), 4 waves/block
    attn_kernel<<<dim3(512 / 64, 32 * 16), 128, 0, stream>>>(Qb, Kb, Vt, Ob);

    // 3) output projection: [16384,1024] x [1024,1024]
    gemm_kernel<1><<<dim3(1024 / BN, 16384 / BM), 256, 0, stream>>>(
        Ob, Wot, b_out, nullptr, nullptr, nullptr, out, 16384, 1024, 1024);
}
